// DGCNN_63093069578680
// MI455X (gfx1250) — matrix operand order repaired and checked by reading, not compiled, not run on hardware
//
#include <hip/hip_runtime.h>

// CDNA5 / gfx1250 DGCNN segmentation forward.
// GEMMs via V_WMMA_F32_16X16X32_F16 (f16 in, f32 accumulate), weights staged in LDS;
// kNN + reductions fp32 VALU. Fragment loaders are branchless; K is compile-time.
// Streaming operands use global_prefetch (gfx1250 __builtin_prefetch).

typedef __attribute__((ext_vector_type(16))) _Float16 v16h;
typedef __attribute__((ext_vector_type(8)))  float    v8f;

#define LSLOPE 0.2f

// ---------------- problem constants (match setup_inputs) ----------------
static constexpr int B_   = 4;
static constexpr int C0_  = 6;
static constexpr int NP_  = 4096;
static constexpr int KNB_ = 20;
static constexpr int NC_  = 8;

// ---------------- workspace layout (float offsets) ----------------
static constexpr size_t XT_OFF    = 0;          //  98304 f
static constexpr size_t IDX_OFF   = 98304;      // 327680 i32
static constexpr size_t X1_OFF    = 425984;     // 1048576 f
static constexpr size_t X2_OFF    = 1474560;
static constexpr size_t X3_OFF    = 2523136;
static constexpr size_t SUM_OFF   = 3571712;    // 1024 f
static constexpr size_t SSQ_OFF   = 3572736;    // 1024 f
static constexpr size_t SCALE_OFF = 3573760;    // 1024 f
static constexpr size_t SHIFT_OFF = 3574784;    // 1024 f
static constexpr size_t GLOB_OFF  = 3575808;    // 4096 f
static constexpr size_t YA_OFF    = 3579904;    // 20971520 f (84MB)
static constexpr size_t YB_OFF    = 24551424;   // 20971520 f (84MB)
static constexpr size_t WS_FLOATS = 45522944;   // ~182MB total

// ---------------- generic WMMA GEMM ----------------
// Y[M,O] = A[M,K] * W[O,K]^T ; A built on the fly per MODE:
//   0: edge features  (gather: [F[j]-F[n], F[n]]), Cin = K/2
//   1: BN+LeakyReLU of previous pre-norm Y buffer (fused activation)
//   2: concat(x1,x2,x3)                      (K = 192)
//   3: concat(glob, x1,x2,x3, x1,x2,x3)      (K = 1408)
// FINAL: add bias and store transposed (B, NCls, Npts) instead of Y; only FINAL
// has O not a multiple of 16, so column guards exist only there.
// Grid: x = M/16/8 (8 M-tiles per block, one per wave), y = O/16 (N-tile).
// The block's 16xK weight tile is staged once into LDS as f16.
struct GP {
  const float* W; float* Y;
  int M, O;
  const float* F; const int* idx;
  const float* Yprev; const float* scale; const float* shift;
  const float* x1; const float* x2; const float* x3; const float* glob;
  const float* bias; float* outT;
};

template <int MODE, bool FINAL, int K>
__global__ __launch_bounds__(256) void k_gemm(GP p) {
  extern __shared__ _Float16 wtile[];            // [16][K] f16 weight tile
  const int tid  = threadIdx.x;
  const int lane = tid & 31;
  const int wave = tid >> 5;
  const int tn   = blockIdx.y;
  const int tm   = blockIdx.x * 8 + wave;

  // ---- stage the 16-output-channel weight tile into LDS (f16) ----
#pragma unroll 1
  for (int ocl = 0; ocl < 16; ++ocl) {
    int oc = tn * 16 + ocl;
    if (FINAL) oc = oc < p.O ? oc : 0;           // O=8: pad with row 0 (never stored)
    const float* wr = p.W + (size_t)oc * K;
    for (int k = tid; k < K; k += 256)
      wtile[ocl * K + k] = (_Float16)wr[k];
  }
  __syncthreads();

  const int rowA = tm * 16 + (lane & 15);        // A: one matrix row per lane

  const float* aC = nullptr; const float* aJ = nullptr; const float* aRow = nullptr;
  const float* x1p = nullptr; const float* x2p = nullptr; const float* x3p = nullptr;
  const float* gp = nullptr;
  constexpr int CIN = K / 2;                     // mode-0 input width
  if (MODE == 0) {
    constexpr int PER = NP_ * KNB_;
    const int b  = rowA / PER;
    const int rm = rowA - b * PER;
    const int n  = rm / KNB_;
    const int kk = rm - n * KNB_;
    const int j  = p.idx[((size_t)b * NP_ + n) * KNB_ + kk];
    aC = p.F + ((size_t)b * NP_ + n) * CIN;
    aJ = p.F + ((size_t)b * NP_ + j) * CIN;
  } else if (MODE == 1) {
    aRow = p.Yprev + (size_t)rowA * K;
  } else {
    x1p = p.x1 + (size_t)rowA * 64;
    x2p = p.x2 + (size_t)rowA * 64;
    x3p = p.x3 + (size_t)rowA * 64;
    if (MODE == 3) gp = p.glob + (size_t)(rowA / NP_) * 1024;
  }

  // Branchless A-element loader: clamped indices + value/pointer selects only.
  auto loadA = [&](int c) -> float {
    if (MODE == 0) {
      int ck = c;
      if ((K & 31) != 0) ck = c < K ? c : K - 1;   // only the K=12 layer needs this
      const int off = ck < CIN ? ck : ck - CIN;
      const float ce = aC[off];
      const float je = aJ[off];
      float v = ck < CIN ? (je - ce) : ce;
      if ((K & 31) != 0) v = c < K ? v : 0.f;
      return v;
    } else if (MODE == 1) {
      const float y = fmaf(aRow[c], p.scale[c], p.shift[c]);
      return fmaxf(y, LSLOPE * y);                 // LeakyReLU, branchless
    } else if (MODE == 2) {
      const float* sp = c < 64 ? (x1p + c) : (c < 128 ? (x2p + c - 64) : (x3p + c - 128));
      return *sp;
    } else {
      const float gv = gp[c < 1024 ? c : 1023];
      int cc = c - 1024; cc = cc < 0 ? 0 : cc;     // keep address valid when c < 1024
      cc = cc < 192 ? cc : cc - 192;               // fold mid duplicate
      const float* sp = cc < 64 ? (x1p + cc) : (cc < 128 ? (x2p + cc - 64) : (x3p + cc - 128));
      const float xv = *sp;
      return c < 1024 ? gv : xv;
    }
  };

  v8f acc = {};
  const int kA8  = (lane >> 4) << 3;   // A: half-group K offsets +0..7/+16..23 vs +8..15/+24..31
  const int kB16 = (lane >> 4) << 4;   // unused: B is loaded with kA8, in the same k order as A
  const _Float16* wlane = wtile + (lane & 15) * K;   // this lane's output channel
#pragma unroll 2
  for (int k0 = 0; k0 < K; k0 += 32) {
    if (MODE == 1 && K > 64 && k0 + 64 < K)
      __builtin_prefetch(aRow + k0 + 64, 0, 1);    // global_prefetch: stream next chunk
    v16h a, bf;
#pragma unroll
    for (int v = 0; v < 16; ++v) {
      const int vg = v >> 1, h = v & 1;
      const int ka = k0 + ((vg & 3) << 1) + h + ((vg >> 2) << 4) + kA8;
      a[v] = (_Float16)loadA(ka);
      int kb = k0 + kA8 + (v & 7) + ((v >> 3) << 4); // same k order as A
      if ((K & 31) != 0) kb = kb < K ? kb : K - 1; // clamp; A zero kills garbage term
      bf[v] = wlane[kb];
    }
    acc = __builtin_amdgcn_wmma_f32_16x16x32_f16(false, a, false, bf,
                                                 (short)0, acc, false, false);
  }

  // D layout: m = v + 8*(lane>=16), n = lane&15
  const int oc = tn * 16 + (lane & 15);
#pragma unroll
  for (int v = 0; v < 8; ++v) {
    const int m = v + ((lane >> 4) << 3);
    const int r = tm * 16 + m;
    if (FINAL) {
      if (oc < p.O) {
        const float val = acc[v] + p.bias[oc];
        const int bb = r / NP_, pt = r - bb * NP_;
        p.outT[((size_t)bb * NC_ + oc) * NP_ + pt] = val;
      }
    } else {
      // O is a multiple of 16 for every non-final layer: no guard needed.
      p.Y[(size_t)r * p.O + oc] = acc[v];
    }
  }
}

// ---------------- transpose (B,C,N) -> (B,N,C) ----------------
__global__ void k_transpose(const float* __restrict__ x, float* __restrict__ xt) {
  const int t = blockIdx.x * blockDim.x + threadIdx.x;
  constexpr int tot = B_ * C0_ * NP_;
  if (t >= tot) return;
  const int n = t % NP_;
  const int rem = t / NP_;
  const int c = rem % C0_;
  const int b = rem / C0_;
  xt[((size_t)b * NP_ + n) * C0_ + c] = x[t];
}

// ---------------- brute-force kNN (k smallest sq dists, tie -> lower index) -----
__global__ __launch_bounds__(256) void k_knn(const float* __restrict__ F, int Cdim,
                                             int* __restrict__ idxOut) {
  __shared__ float q[64];
  __shared__ float dist[NP_];
  __shared__ float rv[256];
  __shared__ int   ri[256];
  const int tid = threadIdx.x;
  const int b = blockIdx.x / NP_;
  const int i = blockIdx.x - b * NP_;
  const float* Fb = F + (size_t)b * NP_ * Cdim;
  if (tid < Cdim) q[tid] = Fb[(size_t)i * Cdim + tid];
  __syncthreads();
  for (int j = tid; j < NP_; j += 256) {
    const float* fj = Fb + (size_t)j * Cdim;
    __builtin_prefetch(fj + 256 * Cdim, 0, 1);
    float d = 0.f;
    for (int c = 0; c < Cdim; ++c) { const float t = fj[c] - q[c]; d += t * t; }
    dist[j] = d;
  }
  __syncthreads();
  for (int it = 0; it < KNB_; ++it) {
    float bv = 3.0e38f; int bi = 0x7fffffff;
    for (int j = tid; j < NP_; j += 256) {
      const float d = dist[j];
      const bool better = (d < bv) || (d == bv && j < bi);
      bv = better ? d : bv;
      bi = better ? j : bi;
    }
    rv[tid] = bv; ri[tid] = bi;
    __syncthreads();
    for (int s = 128; s > 0; s >>= 1) {
      if (tid < s) {
        const float ov = rv[tid + s]; const int oi = ri[tid + s];
        const bool better = (ov < rv[tid]) || (ov == rv[tid] && oi < ri[tid]);
        rv[tid] = better ? ov : rv[tid];
        ri[tid] = better ? oi : ri[tid];
      }
      __syncthreads();
    }
    if (tid == 0) {
      idxOut[((size_t)b * NP_ + i) * KNB_ + it] = ri[0];
      dist[ri[0]] = 3.0e38f;
    }
    __syncthreads();
  }
}

// ---------------- per-channel sum / sumsq over rows ----------------
__global__ void k_stats_reduce(const float* __restrict__ Y, long M, int O,
                               float* __restrict__ sum, float* __restrict__ ssq) {
  const long t = (long)blockIdx.x * blockDim.x + threadIdx.x;
  if (t >= (long)O * 64) return;
  const int o = (int)(t % O);
  const long chunk = t / O;
  float s = 0.f, s2 = 0.f;
  for (long r = chunk; r < M; r += 64) {
    __builtin_prefetch(Y + (size_t)(r + 256) * O + o, 0, 1);
    const float y = Y[(size_t)r * O + o];
    s += y; s2 += y * y;
  }
  atomicAdd(&sum[o], s);
  atomicAdd(&ssq[o], s2);
}

__global__ void k_stats_final(const float* __restrict__ sum, const float* __restrict__ ssq,
                              const float* __restrict__ g, const float* __restrict__ beta,
                              int O, float cnt,
                              float* __restrict__ scale, float* __restrict__ shift) {
  const int t = blockIdx.x * blockDim.x + threadIdx.x;
  if (t >= O) return;
  const float mu  = sum[t] / cnt;
  const float var = ssq[t] / cnt - mu * mu;
  const float sc  = g[t] * rsqrtf(var + 1e-5f);
  scale[t] = sc;
  shift[t] = beta[t] - mu * sc;
}

// ---------------- BN + LReLU + max over neighbors ----------------
__global__ void k_maxk(const float* __restrict__ Y, const float* __restrict__ scale,
                       const float* __restrict__ shift, int O,
                       float* __restrict__ X) {
  const long t = (long)blockIdx.x * blockDim.x + threadIdx.x;
  if (t >= (long)B_ * NP_ * O) return;
  const int o = (int)(t % O);
  const long r = t / O;
  const float* yp = Y + (size_t)r * KNB_ * O + o;
  const float sc = scale[o], sh = shift[o];
  float m = -3.0e38f;
  for (int kk = 0; kk < KNB_; ++kk) {
    __builtin_prefetch(yp + (size_t)(kk + 4) * O, 0, 1);
    const float y = fmaf(yp[(size_t)kk * O], sc, sh);
    m = fmaxf(m, fmaxf(y, LSLOPE * y));
  }
  X[t] = m;
}

// ---------------- BN + LReLU + max over points (global feature) ----------------
__global__ void k_maxn(const float* __restrict__ Y, const float* __restrict__ scale,
                       const float* __restrict__ shift, int O,
                       float* __restrict__ glob) {
  const int t = blockIdx.x * blockDim.x + threadIdx.x;
  if (t >= B_ * O) return;
  const int o = t % O;
  const int b = t / O;
  const float* yp = Y + (size_t)b * NP_ * O + o;
  const float sc = scale[o], sh = shift[o];
  float m = -3.0e38f;
  for (int n = 0; n < NP_; ++n) {
    __builtin_prefetch(yp + (size_t)(n + 8) * O, 0, 1);
    const float y = fmaf(yp[(size_t)n * O], sc, sh);
    m = fmaxf(m, fmaxf(y, LSLOPE * y));
  }
  glob[t] = m;
}

// ---------------- host helpers ----------------
template <int MODE, bool FINAL, int K>
static void launch_gemm(const GP& p, hipStream_t stream) {
  const dim3 grid((p.M / 16) / 8, (p.O + 15) / 16);
  const size_t shm = (size_t)16 * K * sizeof(_Float16);
  k_gemm<MODE, FINAL, K><<<grid, 256, shm, stream>>>(p);
}

static void run_stats(const float* Y, long M, int O, const float* g, const float* beta,
                      float* sum, float* ssq, float* scale, float* shift,
                      hipStream_t stream) {
  hipMemsetAsync(sum, 0, 1024 * sizeof(float), stream);
  hipMemsetAsync(ssq, 0, 1024 * sizeof(float), stream);
  const long thr = (long)O * 64;
  k_stats_reduce<<<(int)((thr + 255) / 256), 256, 0, stream>>>(Y, M, O, sum, ssq);
  k_stats_final<<<(O + 255) / 256, 256, 0, stream>>>(sum, ssq, g, beta, O, (float)M,
                                                     scale, shift);
}

extern "C" void kernel_launch(void* const* d_in, const int* in_sizes, int n_in,
                              void* d_out, int out_size, void* d_ws, size_t ws_size,
                              hipStream_t stream) {
  (void)in_sizes; (void)n_in; (void)out_size;
  if (ws_size < WS_FLOATS * sizeof(float)) return;   // deterministic no-op if ws too small

  const float* x    = (const float*)d_in[0];
  const float* W1   = (const float*)d_in[1];
  const float* g1   = (const float*)d_in[2];
  const float* b1   = (const float*)d_in[3];
  const float* W2   = (const float*)d_in[4];
  const float* g2   = (const float*)d_in[5];
  const float* b2   = (const float*)d_in[6];
  const float* W3   = (const float*)d_in[7];
  const float* g3   = (const float*)d_in[8];
  const float* b3   = (const float*)d_in[9];
  const float* W4   = (const float*)d_in[10];
  const float* g4   = (const float*)d_in[11];
  const float* b4   = (const float*)d_in[12];
  const float* W5   = (const float*)d_in[13];
  const float* g5   = (const float*)d_in[14];
  const float* b5   = (const float*)d_in[15];
  const float* Wg   = (const float*)d_in[16];
  const float* gg   = (const float*)d_in[17];
  const float* bg   = (const float*)d_in[18];
  const float* Wc1  = (const float*)d_in[19];
  const float* gc1  = (const float*)d_in[20];
  const float* bc1  = (const float*)d_in[21];
  const float* Wc2  = (const float*)d_in[22];
  const float* gc2  = (const float*)d_in[23];
  const float* bc2  = (const float*)d_in[24];
  const float* Wc3  = (const float*)d_in[25];
  const float* bias3= (const float*)d_in[26];

  float* ws    = (float*)d_ws;
  float* xt    = ws + XT_OFF;
  int*   idx   = (int*)(ws + IDX_OFF);
  float* x1    = ws + X1_OFF;
  float* x2    = ws + X2_OFF;
  float* x3    = ws + X3_OFF;
  float* sum   = ws + SUM_OFF;
  float* ssq   = ws + SSQ_OFF;
  float* scale = ws + SCALE_OFF;
  float* shift = ws + SHIFT_OFF;
  float* glob  = ws + GLOB_OFF;
  float* Ya    = ws + YA_OFF;
  float* Yb    = ws + YB_OFF;
  float* out   = (float*)d_out;

  const long ME = (long)B_ * NP_ * KNB_;   // 327680 edge rows
  const long MP = (long)B_ * NP_;          // 16384 point rows

  GP p = {};
  p.scale = scale; p.shift = shift; p.idx = idx;
  p.x1 = x1; p.x2 = x2; p.x3 = x3; p.glob = glob;

  // transpose x -> xt (B,N,6)
  {
    constexpr int tot = B_ * C0_ * NP_;
    k_transpose<<<(tot + 255) / 256, 256, 0, stream>>>(x, xt);
  }

  // ---- stage 1 ----
  k_knn<<<B_ * NP_, 256, 0, stream>>>(xt, C0_, idx);
  p.W = W1; p.Y = Ya; p.M = (int)ME; p.O = 64; p.F = xt;
  launch_gemm<0, false, 12>(p, stream);
  run_stats(Ya, ME, 64, g1, b1, sum, ssq, scale, shift, stream);

  p.W = W2; p.Y = Yb; p.Yprev = Ya;
  launch_gemm<1, false, 64>(p, stream);
  run_stats(Yb, ME, 64, g2, b2, sum, ssq, scale, shift, stream);
  k_maxk<<<(int)((MP * 64 + 255) / 256), 256, 0, stream>>>(Yb, scale, shift, 64, x1);

  // ---- stage 2 ----
  k_knn<<<B_ * NP_, 256, 0, stream>>>(x1, 64, idx);
  p.W = W3; p.Y = Ya; p.F = x1;
  launch_gemm<0, false, 128>(p, stream);
  run_stats(Ya, ME, 64, g3, b3, sum, ssq, scale, shift, stream);

  p.W = W4; p.Y = Yb; p.Yprev = Ya;
  launch_gemm<1, false, 64>(p, stream);
  run_stats(Yb, ME, 64, g4, b4, sum, ssq, scale, shift, stream);
  k_maxk<<<(int)((MP * 64 + 255) / 256), 256, 0, stream>>>(Yb, scale, shift, 64, x2);

  // ---- stage 3 ----
  k_knn<<<B_ * NP_, 256, 0, stream>>>(x2, 64, idx);
  p.W = W5; p.Y = Ya; p.F = x2;
  launch_gemm<0, false, 128>(p, stream);
  run_stats(Ya, ME, 64, g5, b5, sum, ssq, scale, shift, stream);
  k_maxk<<<(int)((MP * 64 + 255) / 256), 256, 0, stream>>>(Ya, scale, shift, 64, x3);

  // ---- global embedding: mid(192) @ Wg^T -> Yg (reuse Ya) ----
  p.W = Wg; p.Y = Ya; p.M = (int)MP; p.O = 1024;
  launch_gemm<2, false, 192>(p, stream);
  run_stats(Ya, MP, 1024, gg, bg, sum, ssq, scale, shift, stream);
  k_maxn<<<(B_ * 1024 + 255) / 256, 256, 0, stream>>>(Ya, scale, shift, 1024, glob);

  // ---- head: feats(1408) @ Wc1^T -> Yc1 (reuse Yb) ----
  p.W = Wc1; p.Y = Yb; p.O = 512;
  launch_gemm<3, false, 1408>(p, stream);
  run_stats(Yb, MP, 512, gc1, bc1, sum, ssq, scale, shift, stream);

  // Yc2 = act(Yc1) @ Wc2^T (reuse Ya)
  p.W = Wc2; p.Y = Ya; p.O = 256; p.Yprev = Yb;
  launch_gemm<1, false, 512>(p, stream);
  run_stats(Ya, MP, 256, gc2, bc2, sum, ssq, scale, shift, stream);

  // logits = act(Yc2) @ Wc3^T + bias -> d_out (B, 8, N) transposed store
  p.W = Wc3; p.Y = nullptr; p.O = NC_; p.Yprev = Ya;
  p.bias = bias3; p.outT = out;
  launch_gemm<1, true, 256>(p, stream);
}
